// tryout_51488067944934
// MI455X (gfx1250) — matrix operand order repaired and checked by reading, not compiled, not run on hardware
//
#include <hip/hip_runtime.h>
#include <hip/hip_bf16.h>

// ---------------------------------------------------------------------------
// SimGNN-style graph matcher for MI455X (gfx1250, wave32, WMMA).
// GEMM-shaped work runs on v_wmma_f32_16x16x32_f16 (f16 in, f32 accumulate):
// RGCN relation transforms, Q/K/V projections, flash cross-attention,
// output projection, cosine node_dot.  Each GEMM wave owns a 32x64 C tile
// (8 accumulators) so fragment loads amortize to 1.5 b128/WMMA.
// Irregular work (edge segment-mean, histograms, tiny MLPs) is scalar.
// ---------------------------------------------------------------------------

#define GN    4096      // nodes
#define GE    131072    // edges
#define GIN   128
#define GHID  64
#define GOUT  64
#define GR    12        // relations
#define GNB   15        // bases
#define GH    8         // heads
#define GTD   512       // total dim
#define GK    16        // tensor neurons
#define GBINS 16
#define GNT   4
#define NSLOT 13        // R relations + root slot

typedef __attribute__((ext_vector_type(16))) _Float16 v16h;
typedef __attribute__((ext_vector_type(8)))  _Float16 h8;
typedef __attribute__((ext_vector_type(8)))  float    v8f;

// -------------------------- WMMA fragment helpers --------------------------
// A fragment, 16x32 f16 tile of row-major A[lda], per CDNA5 ISA layout:
// lanes 0-15: row = row0+lane, VGPR0-3 => K 0..7, VGPR4-7 => K 16..23
// lanes16-31: row = row0+lane-16, K 8..15 and K 24..31.
__device__ __forceinline__ v16h load_frag_a(const _Float16* A, int lda,
                                            int row0, int k0) {
  int lane = threadIdx.x & 31;
  int r  = row0 + (lane & 15);
  int kk = k0 + ((lane >> 4) ? 8 : 0);
  const _Float16* p = A + (long long)r * lda + kk;
  h8 lo = *(const h8*)(p);
  h8 hi = *(const h8*)(p + 16);
  v16h f;
#pragma unroll
  for (int i = 0; i < 8; ++i) { f[i] = lo[i]; f[8 + i] = hi[i]; }
  return f;
}

// B fragment for C = A @ Bt^T with Bt stored [N][K] row-major:
// lane = column n, same K order as A: lanes 0-15 K 0..7,16..23; lanes16-31 K 8..15,24..31.
__device__ __forceinline__ v16h load_frag_bt(const _Float16* Bt, int ldb,
                                             int col0, int k0) {
  int lane = threadIdx.x & 31;
  int c  = col0 + (lane & 15);
  int kk = k0 + ((lane >> 4) ? 8 : 0);
  const _Float16* p = Bt + (long long)c * ldb + kk;
  h8 lo = *(const h8*)(p);
  h8 hi = *(const h8*)(p + 16);
  v16h f;
#pragma unroll
  for (int i = 0; i < 8; ++i) { f[i] = lo[i]; f[8 + i] = hi[i]; }
  return f;
}

__device__ __forceinline__ v8f wmma_f16(v16h a, v16h b, v8f c) {
  return __builtin_amdgcn_wmma_f32_16x16x32_f16(false, a, false, b,
                                                (short)0, c, false, false);
}

// ------------------------------- GEMM kernel -------------------------------
// C[M,N] = A[M,K] @ Bt[N,K]^T (+bias[col]).  blockDim=128 (4 waves), each
// wave owns a 32x64 C macro-tile (2 row-frags x 4 col-frags, 8 accums):
// per 32-deep K step: 2 A-frag + 4 B-frag loads feed 8 WMMAs.
// Epilogue can emit f32 C, f16 Ch (same layout), and/or f16 transposed Ct
// ([N,M], ld=M).  blockIdx.z batches Bt (+btz) and C (+cz) for the RGCN
// 13-slot transform.
__global__ void gemm_wmma(const _Float16* __restrict__ A, int lda,
                          const _Float16* __restrict__ Bt, int ldb, long long btz,
                          float* __restrict__ C, int ldc, long long cz,
                          _Float16* __restrict__ Ch, _Float16* __restrict__ Ct,
                          int M, int Nn, int Kk, const float* __restrict__ bias) {
  int wave = threadIdx.x >> 5;
  int row0 = blockIdx.x * 128 + wave * 32;
  int col0 = blockIdx.y * 64;
  if (row0 >= M || col0 >= Nn) return;          // wave-uniform
  Bt += (long long)blockIdx.z * btz;
  if (C) C += (long long)blockIdx.z * cz;
  v8f acc[2][4] = {};
  for (int k0 = 0; k0 < Kk; k0 += 32) {
    if (k0 + 32 < Kk) {                          // gfx1250 global_prefetch_b8
      __builtin_prefetch(A + (long long)row0 * lda + k0 + 32, 0, 1);
      __builtin_prefetch(Bt + (long long)col0 * ldb + k0 + 32, 0, 1);
    }
    v16h a0 = load_frag_a(A, lda, row0,      k0);
    v16h a1 = load_frag_a(A, lda, row0 + 16, k0);
#pragma unroll
    for (int j = 0; j < 4; ++j) {
      v16h b = load_frag_bt(Bt, ldb, col0 + j * 16, k0);
      acc[0][j] = wmma_f16(a0, b, acc[0][j]);
      acc[1][j] = wmma_f16(a1, b, acc[1][j]);
    }
  }
  int lane = threadIdx.x & 31;
  int n = lane & 15, mb = (lane >> 4) * 8;
#pragma unroll
  for (int i = 0; i < 2; ++i)
#pragma unroll
    for (int j = 0; j < 4; ++j) {
      int cc = col0 + j * 16 + n;
      float bv = bias ? bias[cc] : 0.f;
#pragma unroll
      for (int v = 0; v < 8; ++v) {
        int rr = row0 + i * 16 + mb + v;
        float val = acc[i][j][v] + bv;
        if (C)  C[(long long)rr * ldc + cc] = val;
        if (Ch) Ch[(long long)rr * ldc + cc] = (_Float16)val;
        if (Ct) Ct[(long long)cc * M + rr] = (_Float16)val;
      }
    }
}

// --------------------------- Flash attention (1 wave) ----------------------
// Q,K: f16 [N,512] row-major; Vt: f16 [512,N] (head rows h*64..h*64+63).
// One wave per (32 query rows, head): 16 WMMA per 32-key chunk on 8 fragment
// loads; every lane owns one softmax row.  Output written f16 for the
// following out-projection GEMM.
__global__ void attn_wmma(const _Float16* __restrict__ Q,
                          const _Float16* __restrict__ Kt,
                          const _Float16* __restrict__ Vt,
                          _Float16* __restrict__ O, int Nk) {
  int h    = blockIdx.y;
  int row0 = blockIdx.x * 32;
  int lane = threadIdx.x;
  __shared__ float    S[32][32];
  __shared__ _Float16 P[32][32];
  __shared__ float    rAl[32], rMx[32], rSm[32];
  rMx[lane] = -1e30f; rSm[lane] = 0.f;
  __syncthreads();

  const _Float16* Qh = Q  + h * 64;
  const _Float16* Kh = Kt + h * 64;
  const _Float16* Vh = Vt + (long long)h * 64 * Nk;
  v16h q00 = load_frag_a(Qh, GTD, row0,      0);
  v16h q01 = load_frag_a(Qh, GTD, row0,      32);
  v16h q10 = load_frag_a(Qh, GTD, row0 + 16, 0);
  v16h q11 = load_frag_a(Qh, GTD, row0 + 16, 32);
  v8f o[2][4] = {};
  const float scale = 0.125f;                    // 1/sqrt(64)
  int n = lane & 15, mb = (lane >> 4) * 8;

  for (int kc = 0; kc < Nk; kc += 32) {
    v16h b00 = load_frag_bt(Kh, GTD, kc,      0);
    v16h b01 = load_frag_bt(Kh, GTD, kc,      32);
    v16h b10 = load_frag_bt(Kh, GTD, kc + 16, 0);
    v16h b11 = load_frag_bt(Kh, GTD, kc + 16, 32);
    v8f s00 = {}, s01 = {}, s10 = {}, s11 = {};
    s00 = wmma_f16(q00, b00, s00); s00 = wmma_f16(q01, b01, s00);
    s01 = wmma_f16(q00, b10, s01); s01 = wmma_f16(q01, b11, s01);
    s10 = wmma_f16(q10, b00, s10); s10 = wmma_f16(q11, b01, s10);
    s11 = wmma_f16(q10, b10, s11); s11 = wmma_f16(q11, b11, s11);
#pragma unroll
    for (int v = 0; v < 8; ++v) {
      S[mb + v][n]           = s00[v] * scale;
      S[mb + v][16 + n]      = s01[v] * scale;
      S[16 + mb + v][n]      = s10[v] * scale;
      S[16 + mb + v][16 + n] = s11[v] * scale;
    }
    __syncthreads();
    {                                             // one softmax row per lane
      int r = lane;
      float m = rMx[r], mc = m;
      for (int j = 0; j < 32; ++j) mc = fmaxf(mc, S[r][j]);
      float al = __expf(m - mc);
      float sm = rSm[r] * al;
      for (int j = 0; j < 32; ++j) {
        float e = __expf(S[r][j] - mc);
        P[r][j] = (_Float16)e; sm += e;
      }
      rMx[r] = mc; rSm[r] = sm; rAl[r] = al;
    }
    __syncthreads();
#pragma unroll
    for (int i = 0; i < 2; ++i)
#pragma unroll
      for (int v = 0; v < 8; ++v) {               // rescale running O
        float a = rAl[i * 16 + mb + v];
        o[i][0][v] *= a; o[i][1][v] *= a; o[i][2][v] *= a; o[i][3][v] *= a;
      }
    v16h p0 = load_frag_a(&P[0][0], 32, 0,  0);   // P from LDS
    v16h p1 = load_frag_a(&P[0][0], 32, 16, 0);
#pragma unroll
    for (int j = 0; j < 4; ++j) {
      v16h vb = load_frag_bt(Vh, Nk, j * 16, kc);
      o[0][j] = wmma_f16(p0, vb, o[0][j]);
      o[1][j] = wmma_f16(p1, vb, o[1][j]);
    }
    __syncthreads();
  }
#pragma unroll
  for (int i = 0; i < 2; ++i)
#pragma unroll
    for (int v = 0; v < 8; ++v) {
      float inv = 1.f / fmaxf(rSm[i * 16 + mb + v], 1e-30f);
      long long base = (long long)(row0 + i * 16 + mb + v) * GTD + h * 64 + n;
#pragma unroll
      for (int j = 0; j < 4; ++j)
        O[base + j * 16] = (_Float16)(o[i][j][v] * inv);
    }
}

// ----------------------------- scalar kernels ------------------------------
__global__ void k_fill(float* p, long long n, float v) {
  long long i = (long long)blockIdx.x * blockDim.x + threadIdx.x;
  if (i < n) p[i] = v;
}
__global__ void k_f2h(const float* x, _Float16* y, long long n) {
  long long i = (long long)blockIdx.x * blockDim.x + threadIdx.x;
  if (i < n) y[i] = (_Float16)x[i];
}
// Wt[slot][o][i] f16: slots 0..11 = sum_b comp[r,b]*basis[b,i,o]; slot 12 = root^T
__global__ void k_build_wt(const float* comp, const float* basis,
                           const float* root, _Float16* Wt, int din) {
  int idx = blockIdx.x * blockDim.x + threadIdx.x;
  int tot = NSLOT * 64 * din;
  if (idx >= tot) return;
  int slot = idx / (64 * din), rem = idx % (64 * din);
  int o = rem / din, i = rem % din;
  float v;
  if (slot == GR) v = root[i * 64 + o];
  else {
    v = 0.f;
    for (int b = 0; b < GNB; ++b)
      v += comp[slot * GNB + b] * basis[((long long)b * din + i) * 64 + o];
  }
  Wt[idx] = (_Float16)v;
}
__global__ void k_count(const int* dst, const int* et, float* cnt) {
  int e = blockIdx.x * blockDim.x + threadIdx.x;
  if (e >= GE) return;
  atomicAdd(&cnt[dst[e] * GR + et[e]], 1.f);
}
__global__ void k_edge_agg(const int* src, const int* dst, const int* et,
                           const float* cnt, const float* Hm, float* agg) {
  int idx = blockIdx.x * blockDim.x + threadIdx.x;
  int e = idx >> 4;
  if (e >= GE) return;
  int j = (idx & 15) * 4;
  int s = src[e], d = dst[e], r = et[e];
  float nrm = 1.f / fmaxf(cnt[d * GR + r], 1.f);
  const float* hp = Hm + ((long long)s * NSLOT + r) * 64 + j;
  float* ap = agg + (long long)d * 64 + j;
#pragma unroll
  for (int t = 0; t < 4; ++t) atomicAdd(&ap[t], hp[t] * nrm);
}
__global__ void k_combine(const float* agg, const float* Hm, const float* bias,
                          const float* xres, float* xout, int addres, int dorelu) {
  int i = blockIdx.x * blockDim.x + threadIdx.x;
  if (i >= GN * 64) return;
  int nn = i >> 6, o = i & 63;
  float v = agg[i] + Hm[((long long)nn * NSLOT + GR) * 64 + o] + bias[o];
  if (addres) v += xres[i];
  if (dorelu) v = fmaxf(v, 0.f);
  xout[i] = v;
}
__global__ void k_layernorm_h(const float* x, const float* g, const float* b,
                              _Float16* y) {     // f16 out feeds WMMA GEMMs
  int nn = blockIdx.x * blockDim.x + threadIdx.x;
  if (nn >= GN) return;
  const float* xp = x + (long long)nn * 64;
  float mu = 0.f;
  for (int o = 0; o < 64; ++o) mu += xp[o];
  mu *= (1.f / 64.f);
  float var = 0.f;
  for (int o = 0; o < 64; ++o) { float d = xp[o] - mu; var += d * d; }
  var *= (1.f / 64.f);
  float inv = rsqrtf(var + 1e-5f);
  for (int o = 0; o < 64; ++o)
    y[(long long)nn * 64 + o] = (_Float16)((xp[o] - mu) * inv * g[o] + b[o]);
}
__global__ void k_collapse(const float* gca, const float* wh, float* out) {
  int i = blockIdx.x * blockDim.x + threadIdx.x;
  if (i >= GN * 64) return;
  int nn = i >> 6, o = i & 63;
  float v = 0.f;
  for (int h = 0; h < GH; ++h) v += gca[(long long)nn * GTD + h * 64 + o] * wh[h];
  out[i] = v;
}
__global__ void k_colmean(const float* x, float* p) {
  int o = threadIdx.x;
  if (o >= 64) return;
  float s = 0.f;
  for (int nn = 0; nn < GN; ++nn) s += x[(long long)nn * 64 + o];
  p[o] = s / (float)GN;
}
__global__ void k_typesum(const float* x, const int* tl, float* te) {
  int i = blockIdx.x * blockDim.x + threadIdx.x;
  if (i >= GN * 64) return;
  int nn = i >> 6, o = i & 63;
  atomicAdd(&te[tl[nn] * 64 + o], x[i]);
}
__global__ void k_cadot(const float* t1, const float* t2, float* out) {
  int i = threadIdx.x;
  if (i >= 16) return;
  int a = i >> 2, b = i & 3;
  float na = 0.f, nb = 0.f, d = 0.f;
  for (int o = 0; o < 64; ++o) {
    na += t1[a * 64 + o] * t1[a * 64 + o];
    nb += t2[b * 64 + o] * t2[b * 64 + o];
    d  += t1[a * 64 + o] * t2[b * 64 + o];
  }
  out[i] = d / (fmaxf(sqrtf(na), 1e-12f) * fmaxf(sqrtf(nb), 1e-12f));
}
__global__ void k_rownorm_h(const float* x, _Float16* y) {
  int nn = blockIdx.x * blockDim.x + threadIdx.x;
  if (nn >= GN) return;
  float s = 0.f;
  for (int o = 0; o < 64; ++o) s += x[(long long)nn * 64 + o] * x[(long long)nn * 64 + o];
  float inv = 1.f / fmaxf(sqrtf(s), 1e-12f);
  for (int o = 0; o < 64; ++o)
    y[(long long)nn * 64 + o] = (_Float16)(x[(long long)nn * 64 + o] * inv);
}
__device__ __forceinline__ unsigned f2ord(float f) {
  unsigned u = __float_as_uint(f);
  return (u & 0x80000000u) ? ~u : (u | 0x80000000u);
}
__device__ __forceinline__ float ord2f(unsigned u) {
  return (u & 0x80000000u) ? __uint_as_float(u & 0x7fffffffu)
                           : __uint_as_float(~u);
}
__global__ void k_mm_init(unsigned* mm) {
  int i = threadIdx.x;
  if (i < GNT) { mm[2 * i] = 0xFFFFFFFFu; mm[2 * i + 1] = 0u; }
}
__global__ void k_minmax(const float* nd, const int* tl1, const int* tl2,
                         unsigned* mm) {
  long long i = (long long)blockIdx.x * blockDim.x + threadIdx.x;
  if (i >= (long long)GN * GN) return;
  int a = (int)(i >> 12), b = (int)(i & 4095);
  int t = tl1[a];
  if (tl2[b] != t) return;
  unsigned key = f2ord(nd[i]);
  atomicMin(&mm[2 * t], key);
  atomicMax(&mm[2 * t + 1], key);
}
__global__ void k_hist(const float* nd, const int* tl1, const int* tl2,
                       const unsigned* mm, float* hist) {
  long long i = (long long)blockIdx.x * blockDim.x + threadIdx.x;
  if (i >= (long long)GN * GN) return;
  int a = (int)(i >> 12), b = (int)(i & 4095);
  int t = tl1[a];
  if (tl2[b] != t) return;
  float mn = ord2f(mm[2 * t]), mx = ord2f(mm[2 * t + 1]);
  float rng = (mx > mn) ? (mx - mn) : 1.f;
  int bin = (int)floorf((nd[i] - mn) / rng * (float)GBINS);
  bin = min(max(bin, 0), GBINS - 1);
  atomicAdd(&hist[(1 + t) * GBINS + bin], 1.f);
}
__global__ void k_hist0(const float* cad, float* hist) {
  if (threadIdx.x) return;
  float mn = 1e30f, mx = -1e30f;
  for (int i = 0; i < 16; ++i) { mn = fminf(mn, cad[i]); mx = fmaxf(mx, cad[i]); }
  float rng = (mx > mn) ? (mx - mn) : 1.f;
  for (int i = 0; i < 16; ++i) {
    int b = (int)floorf((cad[i] - mn) / rng * (float)GBINS);
    b = min(max(b, 0), GBINS - 1);
    hist[b] += 1.f;
  }
}
__global__ void k_histnorm(const float* hist, float* sim) {
  int i = threadIdx.x;
  if (i >= 5 * GBINS) return;
  int t = i >> 4;
  float s = 0.f;
  for (int j = 0; j < GBINS; ++j) s += hist[t * GBINS + j];
  sim[i] = hist[i] / fmaxf(s, 1.f);
}
__global__ void k_tensornet(const float* p1, const float* p2, const float* W,
                            const float* Wb, const float* b, float* sim80) {
  int k = threadIdx.x;
  if (k >= GK) return;
  float s = 0.f;
  for (int i = 0; i < 64; ++i) {
    float e1 = p1[i];
    for (int j = 0; j < 64; ++j)
      s += e1 * W[((long long)i * 64 + j) * GK + k] * p2[j];
  }
  float blk = 0.f;
  for (int j = 0; j < 64; ++j) blk += Wb[k * 128 + j] * p1[j];
  for (int j = 0; j < 64; ++j) blk += Wb[k * 128 + 64 + j] * p2[j];
  sim80[k] = fmaxf(s + blk + b[k], 0.f);
}
__global__ void k_final(const float* sim, const float* fw, const float* fb,
                        const float* sw, const float* sb, const float* hb,
                        float* out) {
  __shared__ float sv[GK];
  int k = threadIdx.x;
  if (k < GK) {
    float a = fb[k];
    for (int j = 0; j < 96; ++j) a += sim[j] * fw[k * 96 + j];
    sv[k] = tanhf(a);
  }
  __syncthreads();
  if (k == 0) {
    float l = sb[0];
    for (int j = 0; j < GK; ++j) l += sv[j] * sw[j];
    float sc = 1.f / (1.f + __expf(-l));
    out[0] = sc;
    out[1] = sc * hb[0];
  }
}

// ------------------------------- host driver -------------------------------
static inline void* bump(char*& p, size_t bytes) {
  void* r = (void*)p;
  p += (bytes + 255) & ~(size_t)255;
  return r;
}

extern "C" void kernel_launch(void* const* d_in, const int* in_sizes, int n_in,
                              void* d_out, int out_size, void* d_ws, size_t ws_size,
                              hipStream_t stream) {
  (void)in_sizes; (void)n_in; (void)out_size; (void)ws_size;
  // Inputs 0..8 per setup_inputs() order; params pytree leaves in JAX
  // sorted-key order: Wheads, Wk, Wq, Wv, convs[0..3]{basis,bias,comp,root},
  // fc1_b, fc1_w, ln_b, ln_g, mha_in_b, mha_in_w, mha_out_b, mha_out_w,
  // score_b, score_w, tn_W, tn_Wb, tn_b.
  const float* feat[2]  = { (const float*)d_in[0], (const float*)d_in[1] };
  const int*   ei[2]    = { (const int*)d_in[2],   (const int*)d_in[5]   };
  const int*   et[2]    = { (const int*)d_in[3],   (const int*)d_in[6]   };
  const int*   tl[2]    = { (const int*)d_in[4],   (const int*)d_in[7]   };
  const float* hb       = (const float*)d_in[8];
  const float* Wheads   = (const float*)d_in[9];
  const float* Wk       = (const float*)d_in[10];
  const float* Wq       = (const float*)d_in[11];
  const float* Wv       = (const float*)d_in[12];
  const float* cv_basis[4], *cv_bias[4], *cv_comp[4], *cv_root[4];
  for (int l = 0; l < 4; ++l) {
    cv_basis[l] = (const float*)d_in[13 + 4 * l + 0];
    cv_bias [l] = (const float*)d_in[13 + 4 * l + 1];
    cv_comp [l] = (const float*)d_in[13 + 4 * l + 2];
    cv_root [l] = (const float*)d_in[13 + 4 * l + 3];
  }
  const float* fc1_b    = (const float*)d_in[29];
  const float* fc1_w    = (const float*)d_in[30];
  const float* ln_b     = (const float*)d_in[31];
  const float* ln_g     = (const float*)d_in[32];
  const float* mha_in_b = (const float*)d_in[33];
  const float* mha_in_w = (const float*)d_in[34];
  const float* mha_out_b= (const float*)d_in[35];
  const float* mha_out_w= (const float*)d_in[36];
  const float* score_b  = (const float*)d_in[37];
  const float* score_w  = (const float*)d_in[38];
  const float* tn_W     = (const float*)d_in[39];
  const float* tn_Wb    = (const float*)d_in[40];
  const float* tn_b     = (const float*)d_in[41];

  float* out = (float*)d_out;          // [score, pre_ged, ca_dot(16), node_dot(N*N)]
  float* node_dot = out + 18;

  char* wp = (char*)d_ws;
  _Float16* WqH  = (_Float16*)bump(wp, sizeof(_Float16) * GTD * 64);
  _Float16* WkH  = (_Float16*)bump(wp, sizeof(_Float16) * GTD * 64);
  _Float16* WvH  = (_Float16*)bump(wp, sizeof(_Float16) * GTD * 64);
  _Float16* MiH  = (_Float16*)bump(wp, sizeof(_Float16) * 3 * GTD * GTD);
  _Float16* MoH  = (_Float16*)bump(wp, sizeof(_Float16) * GTD * GTD);
  _Float16* WtH  = (_Float16*)bump(wp, sizeof(_Float16) * NSLOT * 128 * 64);
  _Float16* xh   = (_Float16*)bump(wp, sizeof(_Float16) * GN * 128);
  float* Hbuf    = (float*)bump(wp, sizeof(float) * (long long)GN * NSLOT * 64);
  float* cnt     = (float*)bump(wp, sizeof(float) * GN * GR);
  float* agg     = (float*)bump(wp, sizeof(float) * GN * 64);
  float* bufB    = (float*)bump(wp, sizeof(float) * GN * 64);
  float* bufC    = (float*)bump(wp, sizeof(float) * GN * 64);
  _Float16* afh  = (_Float16*)bump(wp, sizeof(_Float16) * GN * 64);
  _Float16* gqh[2], *gkh[2], *gvh[2];
  for (int g = 0; g < 2; ++g) {
    gqh[g] = (_Float16*)bump(wp, sizeof(_Float16) * (long long)GN * GTD);
    gkh[g] = (_Float16*)bump(wp, sizeof(_Float16) * (long long)GN * GTD);
    gvh[g] = (_Float16*)bump(wp, sizeof(_Float16) * (long long)GN * GTD);
  }
  _Float16* qph  = (_Float16*)bump(wp, sizeof(_Float16) * (long long)GN * GTD);
  _Float16* kph  = (_Float16*)bump(wp, sizeof(_Float16) * (long long)GN * GTD);
  _Float16* vth  = (_Float16*)bump(wp, sizeof(_Float16) * (long long)GTD * GN);
  _Float16* aOh  = (_Float16*)bump(wp, sizeof(_Float16) * (long long)GN * GTD);
  float* gca[2]  = { (float*)bump(wp, sizeof(float) * (long long)GN * GTD),
                     (float*)bump(wp, sizeof(float) * (long long)GN * GTD) };
  float* gcac[2] = { (float*)bump(wp, sizeof(float) * GN * 64),
                     (float*)bump(wp, sizeof(float) * GN * 64) };
  _Float16* gn1  = (_Float16*)bump(wp, sizeof(_Float16) * GN * 64);
  _Float16* gn2  = (_Float16*)bump(wp, sizeof(_Float16) * GN * 64);
  float* p1      = (float*)bump(wp, sizeof(float) * 64);
  float* p2      = (float*)bump(wp, sizeof(float) * 64);
  float* te      = (float*)bump(wp, sizeof(float) * 2 * GNT * 64);
  unsigned* mm   = (unsigned*)bump(wp, sizeof(unsigned) * 2 * GNT);
  float* hist    = (float*)bump(wp, sizeof(float) * 5 * GBINS);
  float* sim     = (float*)bump(wp, sizeof(float) * 96);

  const int TB = 256;
  auto blks = [](long long n, int t) { return (unsigned)((n + t - 1) / t); };

  // f16 copies of dense weights
  k_f2h<<<blks(GTD * 64, TB), TB, 0, stream>>>(Wq, WqH, GTD * 64);
  k_f2h<<<blks(GTD * 64, TB), TB, 0, stream>>>(Wk, WkH, GTD * 64);
  k_f2h<<<blks(GTD * 64, TB), TB, 0, stream>>>(Wv, WvH, GTD * 64);
  k_f2h<<<blks(3LL * GTD * GTD, TB), TB, 0, stream>>>(mha_in_w, MiH, 3LL * GTD * GTD);
  k_f2h<<<blks((long long)GTD * GTD, TB), TB, 0, stream>>>(mha_out_w, MoH, (long long)GTD * GTD);

  // ---- RGCN conv stack per graph ----
  for (int g = 0; g < 2; ++g) {
    const float* cur = feat[g];
    int din = GIN;
    for (int l = 0; l < 4; ++l) {
      int tot = NSLOT * 64 * din;
      k_build_wt<<<blks(tot, TB), TB, 0, stream>>>(cv_comp[l], cv_basis[l],
                                                   cv_root[l], WtH, din);
      k_f2h<<<blks((long long)GN * din, TB), TB, 0, stream>>>(cur, xh, (long long)GN * din);
      gemm_wmma<<<dim3(GN / 128, 1, NSLOT), 128, 0, stream>>>(
          xh, din, WtH, din, (long long)64 * din,
          Hbuf, NSLOT * 64, 64, nullptr, nullptr, GN, 64, din, nullptr);
      k_fill<<<blks(GN * GR, TB), TB, 0, stream>>>(cnt, GN * GR, 0.f);
      k_fill<<<blks(GN * 64, TB), TB, 0, stream>>>(agg, GN * 64, 0.f);
      k_count<<<blks(GE, TB), TB, 0, stream>>>(ei[g] + GE, et[g], cnt);
      k_edge_agg<<<blks((long long)GE * 16, TB), TB, 0, stream>>>(
          ei[g], ei[g] + GE, et[g], cnt, Hbuf, agg);
      float* xout = (l & 1) ? bufC : bufB;
      k_combine<<<blks(GN * 64, TB), TB, 0, stream>>>(
          agg, Hbuf, cv_bias[l], cur, xout,
          (l == 1 || l == 2) ? 1 : 0, (l < 3) ? 1 : 0);
      cur = xout;
      din = GHID;
    }
    k_layernorm_h<<<blks(GN, TB), TB, 0, stream>>>(cur, ln_g, ln_b, afh);
    gemm_wmma<<<dim3(GN / 128, GTD / 64, 1), 128, 0, stream>>>(
        afh, 64, WqH, 64, 0, nullptr, GTD, 0, gqh[g], nullptr, GN, GTD, 64, nullptr);
    gemm_wmma<<<dim3(GN / 128, GTD / 64, 1), 128, 0, stream>>>(
        afh, 64, WkH, 64, 0, nullptr, GTD, 0, gkh[g], nullptr, GN, GTD, 64, nullptr);
    gemm_wmma<<<dim3(GN / 128, GTD / 64, 1), 128, 0, stream>>>(
        afh, 64, WvH, 64, 0, nullptr, GTD, 0, gvh[g], nullptr, GN, GTD, 64, nullptr);
  }

  // ---- cross attention, both directions ----
  for (int d = 0; d < 2; ++d) {
    int a = d, b = 1 - d;   // gca[d] = MHA(q from graph d, k/v from other)
    gemm_wmma<<<dim3(GN / 128, GTD / 64, 1), 128, 0, stream>>>(
        gqh[a], GTD, MiH, GTD, 0, nullptr, GTD, 0, qph, nullptr,
        GN, GTD, GTD, mha_in_b);
    gemm_wmma<<<dim3(GN / 128, GTD / 64, 1), 128, 0, stream>>>(
        gkh[b], GTD, MiH + (long long)GTD * GTD, GTD, 0, nullptr, GTD, 0,
        kph, nullptr, GN, GTD, GTD, mha_in_b + GTD);
    gemm_wmma<<<dim3(GN / 128, GTD / 64, 1), 128, 0, stream>>>(
        gvh[b], GTD, MiH + 2LL * GTD * GTD, GTD, 0, nullptr, GTD, 0,
        nullptr, vth, GN, GTD, GTD, mha_in_b + 2 * GTD);   // V written transposed
    attn_wmma<<<dim3(GN / 32, GH), 32, 0, stream>>>(qph, kph, vth, aOh, GN);
    gemm_wmma<<<dim3(GN / 128, GTD / 64, 1), 128, 0, stream>>>(
        aOh, GTD, MoH, GTD, 0, gca[d], GTD, 0, nullptr, nullptr,
        GN, GTD, GTD, mha_out_b);
    k_collapse<<<blks(GN * 64, TB), TB, 0, stream>>>(gca[d], Wheads, gcac[d]);
  }

  // ---- pooled embeddings, tensor net, cosines ----
  k_colmean<<<1, 64, 0, stream>>>(gcac[0], p1);
  k_colmean<<<1, 64, 0, stream>>>(gcac[1], p2);
  k_fill<<<blks(2 * GNT * 64, TB), TB, 0, stream>>>(te, 2 * GNT * 64, 0.f);
  k_typesum<<<blks(GN * 64, TB), TB, 0, stream>>>(gcac[0], tl[0], te);
  k_typesum<<<blks(GN * 64, TB), TB, 0, stream>>>(gcac[1], tl[1], te + GNT * 64);
  k_cadot<<<1, 32, 0, stream>>>(te, te + GNT * 64, out + 2);
  k_rownorm_h<<<blks(GN, TB), TB, 0, stream>>>(gcac[0], gn1);
  k_rownorm_h<<<blks(GN, TB), TB, 0, stream>>>(gcac[1], gn2);
  gemm_wmma<<<dim3(GN / 128, GN / 64, 1), 128, 0, stream>>>(
      gn1, 64, gn2, 64, 0, node_dot, GN, 0, nullptr, nullptr, GN, GN, 64, nullptr);

  // ---- histograms + final score ----
  k_mm_init<<<1, 32, 0, stream>>>(mm);
  k_minmax<<<blks((long long)GN * GN, TB), TB, 0, stream>>>(node_dot, tl[0], tl[1], mm);
  k_fill<<<1, 128, 0, stream>>>(hist, 5 * GBINS, 0.f);
  k_hist0<<<1, 32, 0, stream>>>(out + 2, hist);
  k_hist<<<blks((long long)GN * GN, TB), TB, 0, stream>>>(node_dot, tl[0], tl[1], mm, hist);
  k_histnorm<<<1, 128, 0, stream>>>(hist, sim);
  k_tensornet<<<1, 32, 0, stream>>>(p1, p2, tn_W, tn_Wb, tn_b, sim + 80);
  k_final<<<1, 32, 0, stream>>>(sim, fc1_w, fc1_b, score_w, score_b, hb, out);
}
